// ExplainModule_9869834847153
// MI455X (gfx1250) — matrix operand order repaired and checked by reading, not compiled, not run on hardware
//
#include <hip/hip_runtime.h>
#include <hip/hip_bf16.h>

typedef __attribute__((ext_vector_type(16))) _Float16 v16h;
typedef __attribute__((ext_vector_type(8)))  float    v8f;

// ---------------------------------------------------------------------------
// Kernel 0: prep. Single block, 256 threads.
//  - W1t[col*128 + k] = (f16) W1[k*64 + col]   for k in [0,128), col in [0,64)
//    (B-operand friendly: contiguous K per column)
//  - c64[h] = b1[h] + sum_k embed[node_idx*64+k] * W1[(128+k)*64 + h]
// ---------------------------------------------------------------------------
__global__ __launch_bounds__(256) void prep_kernel(
    const float* __restrict__ embed, const float* __restrict__ W1,
    const float* __restrict__ b1, const int* __restrict__ nidx,
    float* __restrict__ c64, _Float16* __restrict__ W1t)
{
    const int t = threadIdx.x;
    for (int idx = t; idx < 64 * 128; idx += 256) {
        const int colI = idx >> 7;       // 0..63
        const int k    = idx & 127;      // 0..127
        W1t[idx] = (_Float16)W1[k * 64 + colI];
    }
    if (t < 64) {
        const int node = nidx[0];
        const float* se = embed + (long long)node * 64;
        float s = b1[t];
        #pragma unroll 8
        for (int k = 0; k < 64; ++k)
            s += se[k] * W1[(128 + k) * 64 + t];
        c64[t] = s;
    }
}

// ---------------------------------------------------------------------------
// Kernel 1: zero the mask buffer (d_out reused as mask accumulator).
// ---------------------------------------------------------------------------
__global__ __launch_bounds__(256) void zero_kernel(float* __restrict__ p, long long n)
{
    const long long i = ((long long)blockIdx.x * blockDim.x + threadIdx.x) * 4;
    if (i + 3 < n) {
        *(float4*)(p + i) = make_float4(0.f, 0.f, 0.f, 0.f);
    } else {
        for (long long k = i; k < n; ++k) p[k] = 0.f;
    }
}

// ---------------------------------------------------------------------------
// Kernel 2: per-edge MLP via WMMA + gate + atomic scatter into mask.
// One wave32 per 16-edge tile. D = relu(A @ W1[:128] + c) then dot W2.
//   A: 16 edges x 128 (row-part 64 | col-part 64), f16, 4 chunks of K=32
//   B: W1t, f16, per n-tile (4 tiles of 16 hidden cols)
// ---------------------------------------------------------------------------
__global__ __launch_bounds__(256) void gate_scatter_kernel(
    const float* __restrict__ embed, const int* __restrict__ row,
    const int* __restrict__ col, const float* __restrict__ noise,
    const float* __restrict__ W2, const float* __restrict__ b2,
    const float* __restrict__ c64, const _Float16* __restrict__ W1t,
    float* __restrict__ mask, int E_, int N_)
{
    const int lane = threadIdx.x & 31;
    const int tile = blockIdx.x * 8 + (threadIdx.x >> 5);
    if (tile * 16 >= E_) return;                 // uniform per wave -> EXEC stays all-1s

    const int  m    = lane & 15;                 // row (edge-in-tile) / column id
    const bool hi   = lane >= 16;
    const int  base = hi ? 8 : 0;                // table-order K base (unused; A uses khalf)
    const int  khalf = hi ? 16 : 0;              // B-layout K base for this half

    int e = tile * 16 + m;
    if (e >= E_) e = E_ - 1;                     // clamp (E divisible by 16 anyway)
    const int r  = row[e];
    const int cI = col[e];

    // ---- A operand: 4 K-chunks of 32, 16-bit A 16x32, contiguous K (k = 16h + s) ----
    v16h A[4];
    #pragma unroll
    for (int kc = 0; kc < 4; ++kc) {
        const int node = (kc < 2) ? r : cI;
        const float* src = embed + (long long)node * 64 + (kc & 1) * 32 + khalf;
        #pragma unroll
        for (int j = 0; j < 8; ++j) {
            A[kc][j]     = (_Float16)src[j];        // K = k0 + khalf + 0..7
            A[kc][8 + j] = (_Float16)src[8 + j];    // K = k0 + khalf + 8..15
        }
    }

    float cinit[4], w2v[4];
    #pragma unroll
    for (int nt = 0; nt < 4; ++nt) {
        cinit[nt] = c64[nt * 16 + m];            // bias + self-embed contribution
        w2v[nt]   = W2[nt * 16 + m];
    }

    float sum[8];
    #pragma unroll
    for (int v = 0; v < 8; ++v) sum[v] = 0.f;

    #pragma unroll
    for (int nt = 0; nt < 4; ++nt) {
        v8f acc;
        #pragma unroll
        for (int i = 0; i < 8; ++i) acc[i] = cinit[nt];
        #pragma unroll
        for (int kc = 0; kc < 4; ++kc) {
            // B operand: column (nt*16 + m), 16 contiguous K halves (32B aligned)
            const v16h b = *(const v16h*)(W1t + (nt * 16 + m) * 128 + kc * 32 + khalf);
            acc = __builtin_amdgcn_wmma_f32_16x16x32_f16(
                false, A[kc], false, b, (short)0, acc, false, false);
        }
        // relu + second layer partial: acc[v] is (edge v or v+8, hidden nt*16+m)
        #pragma unroll
        for (int v = 0; v < 8; ++v)
            sum[v] += fmaxf(acc[v], 0.f) * w2v[nt];
    }

    const float bias2 = b2[0];
    #pragma unroll
    for (int v = 0; v < 8; ++v) {
        float s = sum[v];
        s += __shfl_xor(s, 1);
        s += __shfl_xor(s, 2);
        s += __shfl_xor(s, 4);
        s += __shfl_xor(s, 8);                   // reduce within each 16-lane half
        if (m == v) {                            // lanes v and 16+v: edges v, v+8
            const int ee = tile * 16 + v + (hi ? 8 : 0);
            if (ee < E_) {
                const float logit = s + bias2;
                const float u = noise[ee];
                const float x = __logf(u) - log1pf(-u) + logit;
                const float g = 1.f / (1.f + __expf(-x));
                atomicAdd(&mask[(long long)row[ee] * N_ + col[ee]], g);
            }
        }
    }
}

// ---------------------------------------------------------------------------
// Kernel 3: in-place symmetrize + apply adj, LDS-tiled transpose (32x32).
// Upper-triangular tile grid; each block handles tiles (ti,tj) and (tj,ti).
// ---------------------------------------------------------------------------
__global__ __launch_bounds__(256) void sym_apply_kernel(
    const float* __restrict__ adj, float* __restrict__ m, int Nn)
{
    const int ti = blockIdx.y, tj = blockIdx.x;
    if (tj < ti) return;
    __shared__ float As[32][33];
    __shared__ float Bs[32][33];
    const int tx = threadIdx.x;      // 0..31
    const int ty = threadIdx.y;      // 0..7
    const int i0 = ti * 32, j0 = tj * 32;

    for (int rr = ty; rr < 32; rr += 8) {
        const int i = i0 + rr, j = j0 + tx;
        As[rr][tx] = (i < Nn && j < Nn) ? m[(long long)i * Nn + j] : 0.f;
        const int i2 = j0 + rr, j2 = i0 + tx;
        Bs[rr][tx] = (i2 < Nn && j2 < Nn) ? m[(long long)i2 * Nn + j2] : 0.f;
    }
    __syncthreads();

    for (int rr = ty; rr < 32; rr += 8) {
        const int i = i0 + rr, j = j0 + tx;
        if (i < Nn && j < Nn) {
            const float s = 0.5f * (As[rr][tx] + Bs[tx][rr]);
            m[(long long)i * Nn + j] = adj[(long long)i * Nn + j] * s;
        }
        if (ti != tj) {
            const int i2 = j0 + rr, j2 = i0 + tx;
            if (i2 < Nn && j2 < Nn) {
                const float s2 = 0.5f * (Bs[rr][tx] + As[tx][rr]);
                m[(long long)i2 * Nn + j2] = adj[(long long)i2 * Nn + j2] * s2;
            }
        }
    }
}

// ---------------------------------------------------------------------------
extern "C" void kernel_launch(void* const* d_in, const int* in_sizes, int n_in,
                              void* d_out, int out_size, void* d_ws, size_t ws_size,
                              hipStream_t stream)
{
    const float* embed = (const float*)d_in[0];
    const int*   row   = (const int*)  d_in[1];
    const int*   col   = (const int*)  d_in[2];
    const float* adj   = (const float*)d_in[3];
    const float* noise = (const float*)d_in[4];
    const float* W1    = (const float*)d_in[5];
    const float* b1    = (const float*)d_in[6];
    const float* W2    = (const float*)d_in[7];
    const float* b2    = (const float*)d_in[8];
    const int*   nidx  = (const int*)  d_in[9];
    float* out = (float*)d_out;

    const int Nn = in_sizes[0] / 64;     // 10000
    const int Ee = in_sizes[1];          // 320000

    // ws layout: [0,256) c64 f32 ; [256, 256+16384) W1t f16
    float*     c64 = (float*)d_ws;
    _Float16*  W1t = (_Float16*)((char*)d_ws + 256);

    // 0) prep: transpose/convert W1, fold self-embedding + b1 into c64
    prep_kernel<<<1, 256, 0, stream>>>(embed, W1, b1, nidx, c64, W1t);

    // 1) zero mask accumulator (d_out)
    const long long NN = (long long)Nn * Nn;
    const long long nThreads = (NN + 3) / 4;
    zero_kernel<<<(unsigned)((nThreads + 255) / 256), 256, 0, stream>>>(out, NN);

    // 2) WMMA edge MLP + gate + atomic scatter
    const int tiles  = (Ee + 15) / 16;
    const int blocks = (tiles + 7) / 8;          // 8 waves (16-edge tiles) per block
    gate_scatter_kernel<<<blocks, 256, 0, stream>>>(
        embed, row, col, noise, W2, b2, c64, W1t, out, Ee, Nn);

    // 3) symmetrize + apply adjacency, in place on d_out
    dim3 g((Nn + 31) / 32, (Nn + 31) / 32), b(32, 8);
    sym_apply_kernel<<<g, b, 0, stream>>>(adj, out, Nn);
}
